// LGNN_69406671503623
// MI455X (gfx1250) — matrix operand order repaired and checked by reading, not compiled, not run on hardware
//
#include <hip/hip_runtime.h>

// ---------------------------------------------------------------------------
// Problem constants (from reference)
// ---------------------------------------------------------------------------
#define NN      6144
#define F0      64
#define F1      128
#define FOUT    16
#define NPLANE  (NN * F1)              // one Tx plane, fp32 elements

// d_out flat offsets (float elements), tuple order:
// x(6144x16), tr_idx(4000), val_idx(1000), prob(6144x6144), last_out(6144x48), x0(6144x64)
#define OFF_X     0
#define OFF_TR    98304
#define OFF_VAL   102304
#define OFF_PROB  103304
#define OFF_LAST  37852040
#define OFF_X0    38146952

#define GEMM_ASYNC (__has_builtin(__builtin_amdgcn_global_load_async_to_lds_b128) && \
                    __has_builtin(__builtin_amdgcn_s_wait_asynccnt))

typedef __attribute__((ext_vector_type(16))) __bf16 v16b;
typedef __attribute__((ext_vector_type(8)))  float  v8f;
typedef __attribute__((ext_vector_type(4)))  int    v4i;
typedef __attribute__((address_space(1))) v4i* as1_v4i_p;
typedef __attribute__((address_space(3))) v4i* as3_v4i_p;

__device__ __forceinline__ unsigned short f2bf(float f) {
    union { float f; unsigned u; } c; c.f = f;
    unsigned r = c.u + 0x7FFFu + ((c.u >> 16) & 1u);   // round-to-nearest-even
    return (unsigned short)(r >> 16);
}

// ---------------------------------------------------------------------------
// h = relu(x5 @ W_gat + b_gat)  (N,6)   and   sq[i] = ||h_i||^2
// ---------------------------------------------------------------------------
__global__ void k_h_sq(const float* __restrict__ x5, const float* __restrict__ Wg,
                       const float* __restrict__ bg, float* __restrict__ h,
                       float* __restrict__ sq) {
    int i = blockIdx.x * blockDim.x + threadIdx.x;
    if (i >= NN) return;
    float a0 = x5[i*3+0], a1 = x5[i*3+1], a2 = x5[i*3+2];
    float s = 0.f;
    #pragma unroll
    for (int c = 0; c < 6; ++c) {
        float v = bg[c] + a0*Wg[0*6+c] + a1*Wg[1*6+c] + a2*Wg[2*6+c];
        v = fmaxf(v, 0.f);
        h[i*6+c] = v;
        s += v*v;
    }
    sq[i] = s;
}

// ---------------------------------------------------------------------------
// Tx0 = h2 = relu(x0 @ W_lin + b_lin)   (N,128)
// ---------------------------------------------------------------------------
__global__ void k_h2(const float* __restrict__ x0, const float* __restrict__ Wl,
                     const float* __restrict__ bl, float* __restrict__ Tx0) {
    int id = blockIdx.x * blockDim.x + threadIdx.x;     // N*128 threads
    int i = id >> 7, f = id & 127;
    float s = bl[f];
    #pragma unroll 8
    for (int d = 0; d < F0; ++d) s += x0[i*F0+d] * Wl[d*F1+f];
    Tx0[id] = fmaxf(s, 0.f);
}

// ---------------------------------------------------------------------------
// prob_matrix row + degree:  P[i][j] = exp(h_i.h_j - 0.5*(sq_i+sq_j))
// deg[i] = rowsum(P) - 1   (diagonal removed; P[i][i] == 1)
// ---------------------------------------------------------------------------
__global__ void k_prob(const float* __restrict__ h, const float* __restrict__ sq,
                       float* __restrict__ P, float* __restrict__ deg) {
    __shared__ float red[256];
    int i = blockIdx.x;
    float h0 = h[i*6+0], h1 = h[i*6+1], h2_ = h[i*6+2];
    float h3 = h[i*6+3], h4 = h[i*6+4], h5 = h[i*6+5];
    float si = sq[i];
    float part = 0.f;
    for (int j = threadIdx.x; j < NN; j += 256) {
        const float* hj = h + j*6;
        float dot = h0*hj[0] + h1*hj[1] + h2_*hj[2] + h3*hj[3] + h4*hj[4] + h5*hj[5];
        float p = expf(dot - 0.5f*(si + sq[j]));
        P[(size_t)i*NN + j] = p;
        part += p;
    }
    red[threadIdx.x] = part;
    __syncthreads();
    for (int s = 128; s > 0; s >>= 1) {
        if (threadIdx.x < s) red[threadIdx.x] += red[threadIdx.x + s];
        __syncthreads();
    }
    if (threadIdx.x == 0) deg[i] = red[0] - 1.0f;
}

__global__ void k_dinv(const float* __restrict__ deg, float* __restrict__ dinv) {
    int i = blockIdx.x * blockDim.x + threadIdx.x;
    if (i >= NN) return;
    float d = deg[i];
    dinv[i] = d > 0.f ? rsqrtf(d) : 0.f;
}

// ---------------------------------------------------------------------------
// S_bf16[i][j] = (i==j) ? 0 : bf16( -dinv_i * P[i][j] * dinv_j )
// ---------------------------------------------------------------------------
__global__ void k_makeS(const float* __restrict__ P, const float* __restrict__ dinv,
                        unsigned short* __restrict__ S) {
    int j = blockIdx.x * 256 + threadIdx.x;
    int i = blockIdx.y;
    float v = -(dinv[i] * P[(size_t)i*NN + j] * dinv[j]);
    S[(size_t)i*NN + j] = (i == j) ? (unsigned short)0 : f2bf(v);
}

// ---------------------------------------------------------------------------
// TxbT[f][j] = bf16(Tx[j][f])   (128 x N, bf16) — B-operand staging
// ---------------------------------------------------------------------------
__global__ void k_tconv(const float* __restrict__ Tx, unsigned short* __restrict__ TxbT) {
    int j = blockIdx.x * 256 + threadIdx.x;
    int f = blockIdx.y;
    TxbT[(size_t)f*NN + j] = f2bf(Tx[(size_t)j*F1 + f]);
}

// ---------------------------------------------------------------------------
// Main WMMA GEMM:  Y = S(bf16, NxN) @ X (via XbT, 128xN bf16), fp32 accumulate.
// Tout = recur ? 2*Y - Tprev : Y      (Chebyshev combine fused in epilogue)
// Wave: 16 rows x 64 cols (4 tiles). Double-buffered K pipeline:
//   A operand: register ping-pong global_load_b128 + global_prefetch
//   B operand: async global->LDS copies (ASYNCcnt), consumed via ds_load_b128
// ---------------------------------------------------------------------------
__global__ void __launch_bounds__(128)
k_gemm(const unsigned short* __restrict__ S, const unsigned short* __restrict__ XbT,
       const float* __restrict__ Tprev, float* __restrict__ Tout, int recur) {
    const int lane = threadIdx.x & 31;
    const int wave = threadIdx.x >> 5;
    const int gw   = blockIdx.x * 4 + wave;
    const int rb   = (gw >> 1) * 16;        // row block base
    const int f00  = (gw & 1) * 64;         // column half base
    const int hi   = lane >> 4;
    const int lo   = lane & 15;

    v8f acc[4];
    #pragma unroll
    for (int t = 0; t < 4; ++t)
        #pragma unroll
        for (int e = 0; e < 8; ++e) acc[t][e] = 0.f;

    // A fragment source: 16x32 bf16 of S, contiguous k = 16*hi + s per lane half (same order as B)
    const unsigned short* Arow = S + (size_t)(rb + lo) * NN + hi * 16;

    union AF { uint4 u[2]; v16b v; };
    AF a0, a1;
    auto loadA = [&](AF& a, int j0) {
        const unsigned short* pa = Arow + j0;
        a.u[0] = *(const uint4*)(pa);        // K 0-7   / 16-23
        a.u[1] = *(const uint4*)(pa + 8);    // K 8-15  / 24-31
    };

#if GEMM_ASYNC
    // per-wave private double buffer: 64 cols x 32 K bf16 = 4KB each
    __shared__ __align__(16) unsigned short smem[4 * 2 * 2048];
    unsigned short* mybuf = smem + wave * 2 * 2048;
    const int crow = lane >> 2;          // 0..7   : row within 8-row stripe
    const int coff = (lane & 3) * 8;     // 0,8,16,24 : element offset (16B chunks)

    auto prefB = [&](int j0, int buf) {
        // fence: LDS reads of this buffer must be complete before async rewrites it
        asm volatile("s_wait_dscnt 0x0" ::: "memory");
        unsigned short* db = mybuf + buf * 2048;
        #pragma unroll
        for (int q = 0; q < 8; ++q) {
            int r = q * 8 + crow;
            const unsigned short* src = XbT + (size_t)(f00 + r) * NN + j0 + coff;
            __builtin_amdgcn_global_load_async_to_lds_b128(
                (as1_v4i_p)src,
                (as3_v4i_p)(db + r * 32 + coff),
                0, 0);
        }
    };

    auto consume = [&](AF& a, int buf) {
        const unsigned short* db = mybuf + buf * 2048;
        #pragma unroll
        for (int t = 0; t < 4; ++t) {
            union BF { uint4 u[2]; v16b v; } b;
            const unsigned short* pb = db + (t * 16 + lo) * 32 + hi * 16;
            b.u[0] = *(const uint4*)(pb);
            b.u[1] = *(const uint4*)(pb + 8);
            acc[t] = __builtin_amdgcn_wmma_f32_16x16x32_bf16(
                         false, a.v, false, b.v, (short)0, acc[t], false, false);
        }
    };

    prefB(0, 0);
    loadA(a0, 0);
    for (int j0 = 0; j0 < NN; j0 += 64) {
        int j2 = (j0 + 64 < NN) ? (j0 + 64) : 0;   // wrap: redundant-but-safe tail
        prefB(j0 + 32, 1);
        loadA(a1, j0 + 32);
        __builtin_prefetch(Arow + j0 + 1024, 0, 1);      // A stream ~16 chunks ahead
        __builtin_amdgcn_s_wait_asynccnt(8);             // chunk j0 resident (8 newer in flight)
        consume(a0, 0);
        prefB(j2, 0);
        loadA(a0, j2);
        __builtin_amdgcn_s_wait_asynccnt(8);             // chunk j0+32 resident
        consume(a1, 1);
    }
#else
    // fallback: register ping-pong for B as well
    union BF { uint4 u[2]; v16b v; };
    BF b0[4], b1[4];
    auto loadB = [&](BF* b, int j0) {
        #pragma unroll
        for (int t = 0; t < 4; ++t) {
            const unsigned short* pb = XbT + (size_t)(f00 + t * 16 + lo) * NN + j0 + hi * 16;
            b[t].u[0] = *(const uint4*)(pb);
            b[t].u[1] = *(const uint4*)(pb + 8);
        }
    };
    auto wmma4 = [&](AF& a, BF* b) {
        #pragma unroll
        for (int t = 0; t < 4; ++t)
            acc[t] = __builtin_amdgcn_wmma_f32_16x16x32_bf16(
                         false, a.v, false, b[t].v, (short)0, acc[t], false, false);
    };
    loadA(a0, 0); loadB(b0, 0);
    for (int j0 = 0; j0 < NN; j0 += 64) {
        int j2 = (j0 + 64 < NN) ? (j0 + 64) : 0;
        loadA(a1, j0 + 32); loadB(b1, j0 + 32);
        __builtin_prefetch(Arow + j0 + 1024, 0, 1);
        wmma4(a0, b0);
        loadA(a0, j2); loadB(b0, j2);
        wmma4(a1, b1);
    }
#endif

    // epilogue: C/D layout, VGPR v -> row v (lanes 0-15) / v+8 (lanes 16-31)
    #pragma unroll
    for (int t = 0; t < 4; ++t) {
        int col = f00 + t * 16 + lo;
        #pragma unroll
        for (int v = 0; v < 8; ++v) {
            int row = rb + v + hi * 8;
            size_t idx = (size_t)row * F1 + col;
            float val = acc[t][v];
            if (recur) val = 2.f * val - Tprev[idx];
            Tout[idx] = val;
        }
    }
}

// ---------------------------------------------------------------------------
// last_out[i][c] : 3 ChebConv projections, c in [0,48)
// ---------------------------------------------------------------------------
__global__ void k_proj(const float* __restrict__ Tx,
                       const float* __restrict__ W3, const float* __restrict__ b3,
                       const float* __restrict__ W6, const float* __restrict__ b6,
                       const float* __restrict__ W9, const float* __restrict__ b9,
                       float* __restrict__ last_out) {
    int id = blockIdx.x * blockDim.x + threadIdx.x;      // N*48 threads
    int i = id / 48, c = id % 48;
    int conv = c >> 4, f = c & 15;
    const float* W; const float* bb; int K;
    if (conv == 0)      { W = W3; bb = b3; K = 3; }
    else if (conv == 1) { W = W6; bb = b6; K = 6; }
    else                { W = W9; bb = b9; K = 9; }
    float s = bb[f];
    for (int k = 0; k < K; ++k) {
        const float* tx = Tx + (size_t)k * NPLANE + (size_t)i * F1;
        const float* w  = W + (size_t)k * F1 * FOUT + f;
        #pragma unroll 8
        for (int d = 0; d < F1; ++d) s += tx[d] * w[d * FOUT];
    }
    last_out[(size_t)i*48 + c] = s;
}

// ---------------------------------------------------------------------------
// x = softmax(last_out @ W_last + b_last)  — one 16-lane group per row
// ---------------------------------------------------------------------------
__global__ void k_softmax(const float* __restrict__ last_out,
                          const float* __restrict__ Wl, const float* __restrict__ bl,
                          float* __restrict__ x) {
    int lane = threadIdx.x & 31;
    int wave = threadIdx.x >> 5;
    int hi = lane >> 4, f = lane & 15;
    int row = blockIdx.x * 16 + wave * 2 + hi;
    const float* lo = last_out + (size_t)row * 48;
    float lg = bl[f];
    #pragma unroll
    for (int c = 0; c < 48; ++c) lg += lo[c] * Wl[c * FOUT + f];
    float m = lg;
    #pragma unroll
    for (int d = 8; d >= 1; d >>= 1) m = fmaxf(m, __shfl_xor(m, d, 16));
    float e = expf(lg - m);
    float s = e;
    #pragma unroll
    for (int d = 8; d >= 1; d >>= 1) s += __shfl_xor(s, d, 16);
    x[(size_t)row * FOUT + f] = e / s;
}

// ---------------------------------------------------------------------------
// passthrough outputs: tr_idx, val_idx (bit-copied), x0
// ---------------------------------------------------------------------------
__global__ void k_copy(const int* __restrict__ tr, const int* __restrict__ val,
                       const float* __restrict__ x0, float* __restrict__ out) {
    int id = blockIdx.x * blockDim.x + threadIdx.x;
    if (id < 4000) {
        ((int*)out)[OFF_TR + id] = tr[id];
    } else if (id < 5000) {
        ((int*)out)[OFF_VAL + (id - 4000)] = val[id - 4000];
    } else {
        int j = id - 5000;
        if (j < NN * F0) out[OFF_X0 + j] = x0[j];
    }
}

// ---------------------------------------------------------------------------
extern "C" void kernel_launch(void* const* d_in, const int* in_sizes, int n_in,
                              void* d_out, int out_size, void* d_ws, size_t ws_size,
                              hipStream_t stream) {
    const float* x5    = (const float*)d_in[0];
    const float* x0    = (const float*)d_in[1];
    const int*   tr    = (const int*)  d_in[2];
    const int*   val   = (const int*)  d_in[3];
    const float* Wg    = (const float*)d_in[4];
    const float* bg    = (const float*)d_in[5];
    const float* Wl    = (const float*)d_in[6];
    const float* bl    = (const float*)d_in[7];
    const float* W3    = (const float*)d_in[8];
    const float* b3    = (const float*)d_in[9];
    const float* W6    = (const float*)d_in[10];
    const float* b6    = (const float*)d_in[11];
    const float* W9    = (const float*)d_in[12];
    const float* b9    = (const float*)d_in[13];
    const float* Wlast = (const float*)d_in[14];
    const float* blast = (const float*)d_in[15];

    float* out = (float*)d_out;

    // workspace carve (all offsets 256B aligned)
    char* w = (char*)d_ws;
    unsigned short* Sb   = (unsigned short*)w;            w += (size_t)NN * NN * 2;       // 75.5 MB
    float*          Tx   = (float*)w;                     w += (size_t)9 * NPLANE * 4;    // 27 MB (9 planes)
    unsigned short* TxbT = (unsigned short*)w;            w += (size_t)F1 * NN * 2;       // 1.5 MB
    float*          h    = (float*)w;                     w += (size_t)NN * 6 * 4;
    float*          sq   = (float*)w;                     w += (size_t)NN * 4;
    float*          deg  = (float*)w;                     w += (size_t)NN * 4;
    float*          dinv = (float*)w;

    float* P        = out + OFF_PROB;
    float* last_out = out + OFF_LAST;
    float* x        = out + OFF_X;

    // front end
    k_h_sq <<<NN/256, 256, 0, stream>>>(x5, Wg, bg, h, sq);
    k_h2   <<<NN*F1/256, 256, 0, stream>>>(x0, Wl, bl, Tx);          // Tx plane 0 = h2
    k_prob <<<NN, 256, 0, stream>>>(h, sq, P, deg);
    k_dinv <<<NN/256, 256, 0, stream>>>(deg, dinv);
    k_makeS<<<dim3(NN/256, NN), 256, 0, stream>>>(P, dinv, Sb);

    // Chebyshev recurrence: Tx1 = S@Tx0 ; Txk = 2*(S@Tx_{k-1}) - Tx_{k-2}
    for (int k = 1; k <= 8; ++k) {
        const float* src  = Tx + (size_t)(k - 1) * NPLANE;
        const float* prev = Tx + (size_t)(k >= 2 ? k - 2 : 0) * NPLANE;
        float*       dst  = Tx + (size_t)k * NPLANE;
        k_tconv<<<dim3(NN/256, F1), 256, 0, stream>>>(src, TxbT);
        k_gemm <<<192, 128, 0, stream>>>(Sb, TxbT, prev, dst, k >= 2 ? 1 : 0);
    }

    // projections + softmax + passthrough
    k_proj   <<<NN*48/256, 256, 0, stream>>>(Tx, W3, b3, W6, b6, W9, b9, last_out);
    k_softmax<<<NN/16, 256, 0, stream>>>(last_out, Wlast, blast, x);
    k_copy   <<<(4000 + 1000 + NN*F0 + 255)/256, 256, 0, stream>>>(tr, val, x0, out);
}
